// FrequencyStyleEncoder_17978733101382
// MI455X (gfx1250) — matrix operand order repaired and checked by reading, not compiled, not run on hardware
//
#include <hip/hip_runtime.h>
#include <cstdint>
#include <cstddef>

#define K_BATCH 64
#define SEQ_L   2048
#define DDIM    128
#define NLAYERS 3
#define MFEAT   256
#define HDIM    512
#define DMODEL  256                 // 2*D
#define KLROWS  (K_BATCH * SEQ_L)   // 131072

typedef __attribute__((ext_vector_type(16))) __bf16 v16bf;
typedef __attribute__((ext_vector_type(8)))  __bf16 v8bf;
typedef __attribute__((ext_vector_type(8)))  float  v8f;

enum { EPI_NONE = 0, EPI_PHI = 1, EPI_DIVROW = 2, EPI_ADDRES = 3, EPI_RELU = 4 };

// Pack two fp32 -> bf16 pair (low = a, high = b). Round-half-up (u + 0x8000)
// then take the high 16 bits of each; the pack is a single V_PERM_B32.
__device__ __forceinline__ unsigned int pack2bf(float a, float b) {
  union { float f; unsigned int u; } ca, cb;
  ca.f = a; cb.f = b;
  unsigned int au = ca.u + 0x8000u;
  unsigned int bu = cb.u + 0x8000u;
#if __has_builtin(__builtin_amdgcn_perm)
  // dst byte0,1 = au byte2,3 ; dst byte2,3 = bu byte2,3
  return __builtin_amdgcn_perm(bu, au, 0x07060302u);
#else
  return (au >> 16) | (bu & 0xFFFF0000u);
#endif
}

__device__ __forceinline__ v16bf cat8(v8bf lo, v8bf hi) {
  return __builtin_shufflevector(lo, hi, 0,1,2,3,4,5,6,7,8,9,10,11,12,13,14,15);
}

// ---------------------------------------------------------------------------
// Generic bf16 WMMA GEMM: C = epi(alpha * op(A) @ op(B)), fp32 accumulate.
// Block tile 128x128, K-step 32, 8 waves (2x4), each wave 64x32 via 4x2 WMMAs.
// transA: A stored (Kdim x Mdim) row-major (A[m,k] = A[k*lda+m])
// transB: B stored (Ndim x Kdim) row-major (B[k,n] = B[n*ldb+k])   (i.e. A@B^T)
// ---------------------------------------------------------------------------
#define BM 128
#define BN 128
#define BK 32
#define LDS_S 40   // padded LDS row stride in bf16 elements (80B, 16B-aligned chunks)

__global__ __launch_bounds__(256) void gemm_wmma(
    const float* __restrict__ A, int lda, long long strideA, int transA,
    const float* __restrict__ B, int ldb, long long strideB, int transB,
    float* __restrict__ C, int ldc, long long strideC,
    int Kdim, float alpha, int epi,
    const float* __restrict__ aux, long long auxStride,
    const float* __restrict__ res, long long resStride)
{
  __shared__ unsigned short As[BM * LDS_S];
  __shared__ unsigned short Bs[BN * LDS_S];

  const int z = blockIdx.z;
  A += (long long)z * strideA;
  B += (long long)z * strideB;
  C += (long long)z * strideC;
  const float* auxp = aux ? aux + (long long)z * auxStride : nullptr;
  const float* resp = res ? res + (long long)z * resStride : nullptr;

  const int m0   = blockIdx.x * BM;
  const int n0   = blockIdx.y * BN;
  const int tid  = threadIdx.x;
  const int lane = tid & 31;
  const int wave = tid >> 5;
  const int wm   = wave >> 2;     // 0..1  (64-row slab)
  const int wn   = wave & 3;      // 0..3  (32-col slab)
  const int lh   = lane >> 4;     // lane half
  const int l16  = lane & 15;

  // ---- hoisted staging addresses (advance by constant stride per K-step) ----
  const float* aSrc; long long aStep; unsigned short* aDst;
  if (!transA) {
    int row = tid >> 1, kb = (tid & 1) * 16;
    aSrc  = A + (long long)(m0 + row) * lda + kb;
    aStep = BK;
    aDst  = &As[row * LDS_S + kb];
  } else {
    int k = (tid >> 4) * 2, mb = (tid & 15) * 8;
    aSrc  = A + (long long)k * lda + m0 + mb;
    aStep = (long long)BK * lda;
    aDst  = &As[mb * LDS_S + k];
  }
  const float* bSrc; long long bStep; unsigned short* bDst;
  if (transB) {
    int nn = tid >> 1, kb = (tid & 1) * 16;
    bSrc  = B + (long long)(n0 + nn) * ldb + kb;
    bStep = BK;
    bDst  = &Bs[nn * LDS_S + kb];
  } else {
    int k = (tid >> 4) * 2, nb = (tid & 15) * 8;
    bSrc  = B + (long long)k * ldb + n0 + nb;
    bStep = (long long)BK * ldb;
    bDst  = &Bs[nb * LDS_S + k];
  }

  v8f acc[4][2];
  #pragma unroll
  for (int i = 0; i < 4; ++i)
    #pragma unroll
    for (int j = 0; j < 2; ++j)
      acc[i][j] = (v8f){0.f,0.f,0.f,0.f,0.f,0.f,0.f,0.f};

  const int nk = Kdim / BK;
  for (int kt = 0; kt < nk; ++kt) {
    // ---- stage A tile (BM x BK) fp32 -> packed bf16 pairs ----
    if (!transA) {
      #pragma unroll
      for (int j = 0; j < 16; j += 2)
        *(unsigned int*)&aDst[j] = pack2bf(aSrc[j], aSrc[j + 1]);
    } else {
      const float* s1 = aSrc + lda;
      #pragma unroll
      for (int j = 0; j < 8; ++j)
        *(unsigned int*)&aDst[j * LDS_S] = pack2bf(aSrc[j], s1[j]);
    }
    // ---- stage B tile (BN cols x BK), stored N-major ----
    if (transB) {
      #pragma unroll
      for (int j = 0; j < 16; j += 2)
        *(unsigned int*)&bDst[j] = pack2bf(bSrc[j], bSrc[j + 1]);
    } else {
      const float* s1 = bSrc + ldb;
      #pragma unroll
      for (int j = 0; j < 8; ++j)
        *(unsigned int*)&bDst[j * LDS_S] = pack2bf(bSrc[j], s1[j]);
    }
    aSrc += aStep;
    bSrc += bStep;
    __syncthreads();

    if (kt + 1 < nk) {  // prefetch next tiles into cache (global_prefetch_b8)
      __builtin_prefetch(aSrc, 0, 1);
      __builtin_prefetch(bSrc, 0, 1);
    }

    // ---- load fragments per documented 16-bit WMMA VGPR layouts ----
    // A 16x32: lanes 0-15 (M=lane): K {0..7,16..23}; lanes 16-31: K {8..15,24..31}
    v16bf af[4];
    #pragma unroll
    for (int i = 0; i < 4; ++i) {
      int row = wm * 64 + i * 16 + l16;
      int sel = lh * 8;
      v8bf lo = *reinterpret_cast<const v8bf*>(&As[row * LDS_S + sel]);
      v8bf hi = *reinterpret_cast<const v8bf*>(&As[row * LDS_S + sel + 16]);
      af[i] = cat8(lo, hi);
    }
    // B 32x16: lane holds one column; lanes 0-15: K {0..7,16..23}; lanes 16-31: K {8..15,24..31}
    v16bf bfr[2];
    #pragma unroll
    for (int j = 0; j < 2; ++j) {
      int coln = wn * 32 + j * 16 + l16;
      int koff = lh * 8;
      v8bf lo = *reinterpret_cast<const v8bf*>(&Bs[coln * LDS_S + koff]);
      v8bf hi = *reinterpret_cast<const v8bf*>(&Bs[coln * LDS_S + koff + 16]);
      bfr[j] = cat8(lo, hi);
    }

    #pragma unroll
    for (int i = 0; i < 4; ++i)
      #pragma unroll
      for (int j = 0; j < 2; ++j)
        acc[i][j] = __builtin_amdgcn_wmma_f32_16x16x32_bf16(
            false, af[i], false, bfr[j], (short)0, acc[i][j], false, false);
    __syncthreads();
  }

  // ---- epilogue + store (C/D layout: VGPR r -> M=r (lanes0-15) / M=r+8) ----
  #pragma unroll
  for (int i = 0; i < 4; ++i) {
    int gmBase = m0 + wm * 64 + i * 16 + lh * 8;
    #pragma unroll
    for (int j = 0; j < 2; ++j) {
      int gn = n0 + wn * 32 + j * 16 + l16;
      #pragma unroll
      for (int r = 0; r < 8; ++r) {
        int gm = gmBase + r;
        float raw = acc[i][j][r];
        float v = raw * alpha;
        if (epi == EPI_PHI)         v = __expf(raw - auxp[gm]) * alpha;
        else if (epi == EPI_DIVROW) v = v / auxp[gm];
        else if (epi == EPI_ADDRES) v = v + resp[(long long)gm * ldc + gn];
        else if (epi == EPI_RELU)   v = fmaxf(v, 0.0f);
        C[(long long)gm * ldc + gn] = v;
      }
    }
  }
}

// ---------------------------------------------------------------------------
// Small support kernels
// ---------------------------------------------------------------------------
__global__ void qr_kernel(const float* __restrict__ pr, const float* __restrict__ pim,
                          float* __restrict__ Ab) {
  __shared__ float r0[128], r1[128];
  int d = threadIdx.x;  // 0..127
  float p0r = pr[d * 2 + 0], p0i = pim[d * 2 + 0];
  float p1r = pr[d * 2 + 1], p1i = pim[d * 2 + 1];
  r0[d] = p0r * p0r + p0i * p0i; __syncthreads();
  for (int s = 64; s; s >>= 1) { if (d < s) r0[d] += r0[d + s]; __syncthreads(); }
  float n0 = sqrtf(r0[0]); __syncthreads();
  float q0r = p0r / n0, q0i = p0i / n0;
  r0[d] = q0r * p1r + q0i * p1i;        // Re<q0,p1>
  r1[d] = q0r * p1i - q0i * p1r;        // Im<q0,p1>
  __syncthreads();
  for (int s = 64; s; s >>= 1) { if (d < s) { r0[d] += r0[d + s]; r1[d] += r1[d + s]; } __syncthreads(); }
  float rr = r0[0], ri = r1[0]; __syncthreads();
  float v1r = p1r - (rr * q0r - ri * q0i);
  float v1i = p1i - (rr * q0i + ri * q0r);
  r0[d] = v1r * v1r + v1i * v1i; __syncthreads();
  for (int s = 64; s; s >>= 1) { if (d < s) r0[d] += r0[d + s]; __syncthreads(); }
  float n1 = sqrtf(r0[0]);
  Ab[d * 4 + 0] = q0r; Ab[d * 4 + 1] = q0i;
  Ab[d * 4 + 2] = v1r / n1; Ab[d * 4 + 3] = v1i / n1;
}

__global__ __launch_bounds__(512) void fft_kernel(const float* __restrict__ seq,
                                                  float* __restrict__ fftb) {
  __shared__ float re[2048], im[2048];
  int kc = blockIdx.x;            // K*2 blocks
  int kb = kc >> 1, c = kc & 1;
  int tid = threadIdx.x;
  const float* src = seq + (long long)kb * SEQ_L * 2 + c;
  for (int l = tid; l < 2048; l += 512) {
    unsigned r = __brev((unsigned)l) >> 21;   // 11-bit reversal
    re[r] = src[(long long)l * 2];
    im[r] = 0.f;
  }
  __syncthreads();
  for (int len = 2; len <= 2048; len <<= 1) {
    int half = len >> 1;
    for (int b = tid; b < 1024; b += 512) {
      int grp = b / half, off = b % half;
      int i0 = grp * len + off, i1 = i0 + half;
      float s, cs;
      __sincosf(-6.283185307179586f * (float)off / (float)len, &s, &cs);
      float xr = re[i1], xi = im[i1];
      float tr = cs * xr - s * xi, ti = cs * xi + s * xr;
      float ur = re[i0], ui = im[i0];
      re[i0] = ur + tr; im[i0] = ui + ti;
      re[i1] = ur - tr; im[i1] = ui - ti;
    }
    __syncthreads();
  }
  float* dst = fftb + ((long long)kb * SEQ_L * 2 + c) * 2;
  for (int l = tid; l < 2048; l += 512) {
    dst[(long long)l * 4 + 0] = re[l];
    dst[(long long)l * 4 + 1] = im[l];
  }
}

__global__ void proj_kernel(const float* __restrict__ fftb, const float* __restrict__ Ab,
                            float* __restrict__ X) {
  long long gid = (long long)blockIdx.x * 256 + threadIdx.x;  // KL*128
  long long n = gid >> 7; int d = (int)(gid & 127);
  const float* f = fftb + n * 4;    // [c=0 re,im, c=1 re,im]
  const float* a = Ab + d * 4;      // [A(d,0) re,im, A(d,1) re,im]
  float zr = f[0] * a[0] - f[1] * a[1] + f[2] * a[2] - f[3] * a[3];
  float zi = f[0] * a[1] + f[1] * a[0] + f[2] * a[3] + f[3] * a[2];
  X[n * DMODEL + d]        = zr;
  X[n * DMODEL + DDIM + d] = zi;
}

__global__ void rownorm_kernel(const float* __restrict__ Q, float* __restrict__ rn) {
  int row = blockIdx.x * 8 + (threadIdx.x >> 5);
  int lane = threadIdx.x & 31;
  const float* p = Q + (long long)row * DMODEL;
  float s = 0.f;
  for (int c = lane; c < DMODEL; c += 32) { float v = p[c]; s += v * v; }
  for (int off = 16; off; off >>= 1) s += __shfl_xor(s, off, 32);
  if (lane == 0) rn[row] = 0.5f * s;
}

__global__ void pksum_kernel(const float* __restrict__ pk, float* __restrict__ pks) {
  int kb = blockIdx.x; int m = threadIdx.x;
  const float* p = pk + (long long)kb * SEQ_L * MFEAT + m;
  float s = 0.f;
  for (int l = 0; l < SEQ_L; ++l) s += p[(long long)l * MFEAT];
  pks[kb * MFEAT + m] = s;
}

__global__ void den_kernel(const float* __restrict__ pq, const float* __restrict__ pks,
                           float* __restrict__ den) {
  int row = blockIdx.x * 8 + (threadIdx.x >> 5);   // [0, KL)
  int lane = threadIdx.x & 31;
  int kb = row / SEQ_L;
  const float* p = pq + (long long)row * MFEAT;
  const float* s0 = pks + kb * MFEAT;
  float s = 0.f;
  for (int m = lane; m < MFEAT; m += 32) s += p[m] * s0[m];
  for (int off = 16; off; off >>= 1) s += __shfl_xor(s, off, 32);
  if (lane == 0) den[row] = s + 1e-6f;
}

__global__ void modrelu_kernel(float* __restrict__ X, const float* __restrict__ b) {
  long long gid = (long long)blockIdx.x * 256 + threadIdx.x;  // KL*128
  long long n = gid >> 7; int d = (int)(gid & 127);
  float* row = X + n * DMODEL;
  float zr = row[d], zi = row[d + DDIM];
  float mag = sqrtf(zr * zr + zi * zi);
  float f = fmaxf(mag + b[d], 0.f) / (mag + 1e-6f);
  row[d] = f * zr; row[d + DDIM] = f * zi;
}

// WT1[n][k] = Wc1[k][n], Wc1 (2D x 2H) = [[W1r, W1i], [-W1i, W1r]]
__global__ void pack_w1(const float* __restrict__ w1r, const float* __restrict__ w1i,
                        float* __restrict__ WT) {
  int gid = blockIdx.x * 256 + threadIdx.x;   // 1024*256
  int n = gid >> 8, k = gid & 255;
  int kk = k & (DDIM - 1), nn = n & (HDIM - 1);
  float v;
  if (k < DDIM) v = (n < HDIM) ?  w1r[kk * HDIM + nn] : w1i[kk * HDIM + nn];
  else          v = (n < HDIM) ? -w1i[kk * HDIM + nn] : w1r[kk * HDIM + nn];
  WT[n * (2 * DDIM) + k] = v;
}

// WT2[n][k] = Wc2[k][n], Wc2 (2H x 2D) = [[W2r, W2i], [-W2i, W2r]]
__global__ void pack_w2(const float* __restrict__ w2r, const float* __restrict__ w2i,
                        float* __restrict__ WT) {
  int gid = blockIdx.x * 256 + threadIdx.x;   // 256*1024
  int n = gid >> 10, k = gid & 1023;
  int kk = k & (HDIM - 1), nn = n & (DDIM - 1);
  float v;
  if (k < HDIM) v = (n < DDIM) ?  w2r[kk * DDIM + nn] : w2i[kk * DDIM + nn];
  else          v = (n < DDIM) ? -w2i[kk * DDIM + nn] : w2r[kk * DDIM + nn];
  WT[n * (2 * HDIM) + k] = v;
}

__global__ void abs_kernel(const float* __restrict__ X, float* __restrict__ out) {
  long long gid = (long long)blockIdx.x * 256 + threadIdx.x;  // KL*128
  long long n = gid >> 7; int d = (int)(gid & 127);
  float zr = X[n * DMODEL + d], zi = X[n * DMODEL + DDIM + d];
  out[n * DDIM + d] = sqrtf(zr * zr + zi * zi);
}

// ---------------------------------------------------------------------------
extern "C" void kernel_launch(void* const* d_in, const int* in_sizes, int n_in,
                              void* d_out, int out_size, void* d_ws, size_t ws_size,
                              hipStream_t stream) {
  const float* seq     = (const float*)d_in[0];
  const float* proj_re = (const float*)d_in[1];
  const float* proj_im = (const float*)d_in[2];
  const float* Wq      = (const float*)d_in[3];
  const float* Wk      = (const float*)d_in[4];
  const float* Wv      = (const float*)d_in[5];
  const float* Wo      = (const float*)d_in[6];
  const float* omega   = (const float*)d_in[7];
  const float* attn_b  = (const float*)d_in[8];
  const float* W1_re   = (const float*)d_in[9];
  const float* W1_im   = (const float*)d_in[10];
  const float* W2_re   = (const float*)d_in[11];
  const float* W2_im   = (const float*)d_in[12];
  const float* ffn_b   = (const float*)d_in[13];
  float* out = (float*)d_out;

  float* ws = (float*)d_ws;
  size_t off = 0;
  const size_t ACT = (size_t)KLROWS * DMODEL;           // 33.55M floats
  float* fftb = ws + off; off += (size_t)KLROWS * 4;    // (K,L,2) complex
  float* Ab   = ws + off; off += 512;                   // (D,2) complex
  float* X    = ws + off; off += ACT;
  float* B0   = ws + off; off += ACT;                   // Q / OUT / FFN-hidden part
  float* B1   = ws + off; off += ACT;                   // Kn
  float* B2   = ws + off; off += ACT;                   // V
  float* B3   = ws + off; off += ACT;                   // pq
  float* B4   = ws + off; off += ACT;                   // pk
  float* kvb  = ws + off; off += (size_t)K_BATCH * MFEAT * DMODEL;
  float* WT1  = ws + off; off += (size_t)2 * HDIM * 2 * DDIM;
  float* WT2  = ws + off; off += (size_t)2 * DDIM * 2 * HDIM;
  float* rnQ  = ws + off; off += KLROWS;
  float* rnK  = ws + off; off += KLROWS;
  float* pks  = ws + off; off += (size_t)K_BATCH * MFEAT;
  float* denv = ws + off; off += KLROWS;
  float* Hb   = B0;   // (KL x 1024) aliases contiguous B0..B3
  (void)ws_size; (void)n_in; (void)in_sizes; (void)out_size;

  auto gemm = [&](const float* A, int lda, long long sA, int tA,
                  const float* B, int ldb, long long sB, int tB,
                  float* C, int ldc, long long sC,
                  int M, int N, int Kd, int batch, float alpha, int epi,
                  const float* aux, long long auxS, const float* res) {
    dim3 g(M / BM, N / BN, batch);
    gemm_wmma<<<g, 256, 0, stream>>>(A, lda, sA, tA, B, ldb, sB, tB,
                                     C, ldc, sC, Kd, alpha, epi, aux, auxS, res, sC);
  };

  // --- frontend: QR, FFT, projection ---
  qr_kernel<<<1, 128, 0, stream>>>(proj_re, proj_im, Ab);
  fft_kernel<<<K_BATCH * 2, 512, 0, stream>>>(seq, fftb);
  proj_kernel<<<(KLROWS * DDIM) / 256, 256, 0, stream>>>(fftb, Ab, X);

  const float s_scale = 0.25f;      // DM^-0.25, DM=256
  const float inv_sqrt_m = 0.0625f; // 1/sqrt(256)

  for (int l = 0; l < NLAYERS; ++l) {
    const float* wq  = Wq + (size_t)l * DMODEL * DMODEL;
    const float* wk  = Wk + (size_t)l * DMODEL * DMODEL;
    const float* wv  = Wv + (size_t)l * DMODEL * DMODEL;
    const float* wo  = Wo + (size_t)l * DMODEL * DMODEL;
    const float* om  = omega + (size_t)l * MFEAT * DMODEL;
    const float* w1r = W1_re + (size_t)l * DDIM * HDIM;
    const float* w1i = W1_im + (size_t)l * DDIM * HDIM;
    const float* w2r = W2_re + (size_t)l * HDIM * DDIM;
    const float* w2i = W2_im + (size_t)l * HDIM * DDIM;

    // qn = s * X @ Wq^T ; kn = s * X @ Wk^T ; v = X @ Wv^T
    gemm(X, DMODEL, 0, 0, wq, DMODEL, 0, 1, B0, DMODEL, 0,
         KLROWS, DMODEL, DMODEL, 1, s_scale, EPI_NONE, nullptr, 0, nullptr);
    gemm(X, DMODEL, 0, 0, wk, DMODEL, 0, 1, B1, DMODEL, 0,
         KLROWS, DMODEL, DMODEL, 1, s_scale, EPI_NONE, nullptr, 0, nullptr);
    gemm(X, DMODEL, 0, 0, wv, DMODEL, 0, 1, B2, DMODEL, 0,
         KLROWS, DMODEL, DMODEL, 1, 1.0f, EPI_NONE, nullptr, 0, nullptr);

    rownorm_kernel<<<KLROWS / 8, 256, 0, stream>>>(B0, rnQ);
    rownorm_kernel<<<KLROWS / 8, 256, 0, stream>>>(B1, rnK);

    // phi: exp(u @ om^T - 0.5||u||^2) / sqrt(M)
    gemm(B0, DMODEL, 0, 0, om, DMODEL, 0, 1, B3, MFEAT, 0,
         KLROWS, MFEAT, DMODEL, 1, inv_sqrt_m, EPI_PHI, rnQ, 0, nullptr);
    gemm(B1, DMODEL, 0, 0, om, DMODEL, 0, 1, B4, MFEAT, 0,
         KLROWS, MFEAT, DMODEL, 1, inv_sqrt_m, EPI_PHI, rnK, 0, nullptr);

    pksum_kernel<<<K_BATCH, MFEAT, 0, stream>>>(B4, pks);

    // kv[k] = pk[k]^T @ V[k]   (batched TN, M=256,N=256,K=2048)
    gemm(B4, MFEAT, (long long)SEQ_L * MFEAT, 1,
         B2, DMODEL, (long long)SEQ_L * DMODEL, 0,
         kvb, DMODEL, (long long)MFEAT * DMODEL,
         MFEAT, DMODEL, SEQ_L, K_BATCH, 1.0f, EPI_NONE, nullptr, 0, nullptr);

    den_kernel<<<KLROWS / 8, 256, 0, stream>>>(B3, pks, denv);

    // out[k] = (pq[k] @ kv[k]) / den   (batched NN, M=2048,N=256,K=256)
    gemm(B3, MFEAT, (long long)SEQ_L * MFEAT, 0,
         kvb, DMODEL, (long long)MFEAT * DMODEL, 0,
         B0, DMODEL, (long long)SEQ_L * DMODEL,
         SEQ_L, DMODEL, MFEAT, K_BATCH, 1.0f, EPI_DIVROW, denv, SEQ_L, nullptr);

    // x = x + out @ Wo^T ; modrelu
    gemm(B0, DMODEL, 0, 0, wo, DMODEL, 0, 1, X, DMODEL, 0,
         KLROWS, DMODEL, DMODEL, 1, 1.0f, EPI_ADDRES, nullptr, 0, X);
    modrelu_kernel<<<(KLROWS * DDIM) / 256, 256, 0, stream>>>(X, attn_b + (size_t)l * DDIM);

    // complex FFN via realified block weights
    pack_w1<<<(2 * HDIM * 2 * DDIM) / 256, 256, 0, stream>>>(w1r, w1i, WT1);
    pack_w2<<<(2 * DDIM * 2 * HDIM) / 256, 256, 0, stream>>>(w2r, w2i, WT2);

    gemm(X, DMODEL, 0, 0, WT1, 2 * DDIM, 0, 1, Hb, 2 * HDIM, 0,
         KLROWS, 2 * HDIM, DMODEL, 1, 1.0f, EPI_RELU, nullptr, 0, nullptr);
    gemm(Hb, 2 * HDIM, 0, 0, WT2, 2 * HDIM, 0, 1, X, DMODEL, 0,
         KLROWS, DMODEL, 2 * HDIM, 1, 1.0f, EPI_ADDRES, nullptr, 0, X);
    modrelu_kernel<<<(KLROWS * DDIM) / 256, 256, 0, stream>>>(X, ffn_b + (size_t)l * DDIM);
  }

  abs_kernel<<<(KLROWS * DDIM) / 256, 256, 0, stream>>>(X, out);
}
